// TGNN_52913997086944
// MI455X (gfx1250) — matrix operand order repaired and checked by reading, not compiled, not run on hardware
//
#include <hip/hip_runtime.h>
#include <hip/hip_bf16.h>

// ---------------------------------------------------------------------------
// Problem constants (from reference): NAF=64, NBF=32, IN=160, GROUP=100,
// B=2000, P=200000.  Pair MLP: 160 -> (128|128|128)=384 -> 512 -> 256 -> 1024
// segment-mean(100) -> 1024 -> 512 -> 32 -> 1
// ---------------------------------------------------------------------------
#define GROUP     100
#define IN_FEAT   160
#define CHUNK_M   32          // rows per workgroup chunk (2 WMMA M-tiles)
#define CHUNKS    4           // ceil(100/32)

typedef __attribute__((ext_vector_type(16))) _Float16 v16h;
typedef __attribute__((ext_vector_type(8)))  _Float16 v8h;
typedef __attribute__((ext_vector_type(8)))  float    v8f;

// ---------------------------------------------------------------------------
// WMMA helpers (CDNA5 wave32, V_WMMA_F32_16X16X32_F16)
// ---------------------------------------------------------------------------
__device__ __forceinline__ v8f wmma_f16(v16h a, v16h b, v8f c) {
  // 8 args: (neg_a, A, neg_b, B, c_mod, C, reuse_a, reuse_b)
  return __builtin_amdgcn_wmma_f32_16x16x32_f16(false, a, false, b,
                                                (short)0, c, false, false);
}

// A fragment (16x32 f16) from LDS row-major [M, strideK] at (mBase, kBase).
// Layout: lane L -> row M = L&15; hi = L>>4; K = 16*hi + [0,16) contiguous
// (matches load_b_fragT).
__device__ __forceinline__ v16h load_a_frag(const _Float16* lds, int strideK,
                                            int mBase, int kBase, int lane) {
  int m  = mBase + (lane & 15);
  int hi = lane >> 4;
  const _Float16* p0 = lds + (size_t)m * strideK + kBase + 16 * hi;
  v8h c0 = *(const v8h*)p0;         // 16B LDS load
  v8h c1 = *(const v8h*)(p0 + 8);   // 16B LDS load
  v16h r;
#pragma unroll
  for (int i = 0; i < 8; ++i) { r[i] = c0[i]; r[i + 8] = c1[i]; }
  return r;
}

// B fragment (32x16 f16) from TRANSPOSED weight wT[N][K] (row-major, stride K)
// at tile (nBase, kBase).  Layout (this code's order): lane L -> col N = L&15; K = 16*(L>>4)
// + [0,16) contiguous -> one 32-byte load.
__device__ __forceinline__ v16h load_b_fragT(const _Float16* wT, int K,
                                             int nBase, int kBase, int lane) {
  int n  = nBase + (lane & 15);
  int k0 = kBase + 16 * (lane >> 4);
  return *(const v16h*)(wT + (size_t)n * K + k0);
}

// Dual-M-tile GEMM job: rows [0,32) of aLds against one 16-wide N tile of wT.
// Each B fragment is loaded ONCE and feeds TWO WMMAs (register blocking),
// halving global B traffic and giving the scheduler independent MMA work.
__device__ __forceinline__ void gemm_2m(const _Float16* aLds, int aStride,
                                        int aOff, const _Float16* wT, int kTot,
                                        int nBase, v8f& acc0, v8f& acc1,
                                        int lane) {
  for (int k = 0; k < kTot; k += 32) {
    v16h b  = load_b_fragT(wT, kTot, nBase, k, lane);
    v16h a0 = load_a_frag(aLds, aStride, 0,  aOff + k, lane);
    v16h a1 = load_a_frag(aLds, aStride, 16, aOff + k, lane);
    acc0 = wmma_f16(a0, b, acc0);
    acc1 = wmma_f16(a1, b, acc1);
  }
}

// C/D tile store with bias + ReLU into f16 LDS [M, strideN].
// Lane L holds col N = L&15, rows M = r + 8*(L>>4), r=0..7.
__device__ __forceinline__ void store_relu_f16(_Float16* dst, int strideN,
                                               int mBase, int nBase, v8f acc,
                                               const float* bias, int lane) {
  int n  = lane & 15;
  int hi = lane >> 4;
  float bv = bias[n];
#pragma unroll
  for (int r = 0; r < 8; ++r) {
    float v = acc[r] + bv;
    dst[(size_t)(mBase + r + 8 * hi) * strideN + nBase + n] =
        (_Float16)(v > 0.f ? v : 0.f);
  }
}

// ---------------------------------------------------------------------------
// Kernel 0: zero the per-group accumulator
// ---------------------------------------------------------------------------
__global__ void tgnn_zero_kernel(float* p, int n) {
  int i = blockIdx.x * blockDim.x + threadIdx.x;
  if (i < n) p[i] = 0.f;
}

// ---------------------------------------------------------------------------
// Kernel 1: weight prep — f32 [K,N] -> f16 transposed [N,K]
// ---------------------------------------------------------------------------
__global__ void tgnn_wprep_kernel(const float* __restrict__ w,
                                  _Float16* __restrict__ wT, int K, int N) {
  int i = blockIdx.x * blockDim.x + threadIdx.x;
  if (i >= K * N) return;
  int n = i / K, k = i % K;
  wT[i] = (_Float16)w[(size_t)k * N + n];
}

// ---------------------------------------------------------------------------
// Kernel 2: fused pair MLP + masked column sums into hacc[B][1024]
// grid = B*CHUNKS blocks, 256 threads (8 waves)
// LDS overlay: regX = max(ldsIn 32x160, h1 32x512) = 32768 B
//              regY = max(h384 32x384, h2 32x256)  = 24576 B
// ---------------------------------------------------------------------------
__global__ __launch_bounds__(256) void tgnn_pair_mlp_kernel(
    const float* __restrict__ pairs,
    const float* __restrict__ ab,  const float* __restrict__ bb,
    const float* __restrict__ p1b, const float* __restrict__ p2b,
    const _Float16* __restrict__ awT,  const _Float16* __restrict__ bwT,
    const _Float16* __restrict__ p1wT, const _Float16* __restrict__ p2wT,
    const _Float16* __restrict__ p3wT,
    float* __restrict__ hacc) {
  __shared__ __align__(16) char smem[32768 + 24576];
  _Float16* regX = (_Float16*)smem;
  _Float16* regY = (_Float16*)(smem + 32768);

  const int blk   = blockIdx.x;
  const int group = blk >> 2;     // / CHUNKS
  const int chunk = blk & 3;
  const int row0  = group * GROUP + chunk * CHUNK_M;
  const int tid   = threadIdx.x;
  const int lane  = tid & 31;
  const int wave  = tid >> 5;

  // ---- Stage 0: load pair rows -> f16 LDS [32][160], zero-pad tail rows ----
  _Float16* ldsIn = regX;
  for (int i = tid; i < CHUNK_M * IN_FEAT; i += 256) {
    int r = i / IN_FEAT, c = i % IN_FEAT;
    float v = 0.f;
    if (chunk * CHUNK_M + r < GROUP)
      v = pairs[(size_t)(row0 + r) * IN_FEAT + c];
    ldsIn[i] = (_Float16)v;
  }
  __syncthreads();

  // ---- Stage 1: h384[32][384] = relu([a1@aw | a2@aw | b@bw] + bias) --------
  // 24 N-tile jobs, 2 M-tiles each (register blocked)
  _Float16* h384 = regY;
  for (int nt = wave; nt < 24; nt += 8) {
    const _Float16* wT; const float* bias; int kIn, inOff, wN;
    if (nt < 16) { wT = awT; bias = ab; kIn = 64; inOff = (nt < 8) ? 0 : 64; wN = (nt & 7) * 16; }
    else         { wT = bwT; bias = bb; kIn = 32; inOff = 128;               wN = (nt - 16) * 16; }
    v8f acc0 = {}, acc1 = {};
    gemm_2m(ldsIn, IN_FEAT, inOff, wT, kIn, wN, acc0, acc1, lane);
    store_relu_f16(h384, 384, 0,  nt * 16, acc0, bias + wN, lane);
    store_relu_f16(h384, 384, 16, nt * 16, acc1, bias + wN, lane);
  }
  __syncthreads();

  // ---- Stage 2: h1[32][512] = relu(h384 @ p1w + p1b) -----------------------
  _Float16* h1 = regX;   // ldsIn dead
  for (int nt = wave; nt < 32; nt += 8) {
    v8f acc0 = {}, acc1 = {};
    gemm_2m(h384, 384, 0, p1wT, 384, nt * 16, acc0, acc1, lane);
    store_relu_f16(h1, 512, 0,  nt * 16, acc0, p1b + nt * 16, lane);
    store_relu_f16(h1, 512, 16, nt * 16, acc1, p1b + nt * 16, lane);
  }
  __syncthreads();

  // ---- Stage 3: h2[32][256] = relu(h1 @ p2w + p2b) -------------------------
  _Float16* h2 = regY;   // h384 dead
  for (int nt = wave; nt < 16; nt += 8) {
    v8f acc0 = {}, acc1 = {};
    gemm_2m(h1, 512, 0, p2wT, 512, nt * 16, acc0, acc1, lane);
    store_relu_f16(h2, 256, 0,  nt * 16, acc0, p2b + nt * 16, lane);
    store_relu_f16(h2, 256, 16, nt * 16, acc1, p2b + nt * 16, lane);
  }
  __syncthreads();

  // ---- Stage 4: x = h2 @ p3w ; masked column sums -> atomicAdd hacc --------
  // (p3 bias is added once per group in the head kernel: mean(x+b)=mean+b)
  const int validRows = min(GROUP - chunk * CHUNK_M, CHUNK_M);  // 32,32,32,4
  for (int nt = wave; nt < 64; nt += 8) {
    v8f acc0 = {}, acc1 = {};
    gemm_2m(h2, 256, 0, p3wT, 256, nt * 16, acc0, acc1, lane);
    int hi = lane >> 4;
    float colSum = 0.f;
#pragma unroll
    for (int r = 0; r < 8; ++r) {
      int m0 = r + 8 * hi;          // rows of M-tile 0
      int m1 = 16 + r + 8 * hi;     // rows of M-tile 1
      if (m0 < validRows) colSum += acc0[r];
      if (m1 < validRows) colSum += acc1[r];
    }
    // lane n and lane n+16 cover the two M-halves of column n: combine.
    colSum += __shfl_xor(colSum, 16, 32);
    if (lane < 16)
      atomicAdd(&hacc[(size_t)group * 1024 + nt * 16 + lane], colSum);
  }
}

// ---------------------------------------------------------------------------
// Kernel 3: head MLP over B rows, 16 rows per workgroup (125 blocks)
// ---------------------------------------------------------------------------
__global__ __launch_bounds__(256) void tgnn_head_kernel(
    const float* __restrict__ hacc, const int* __restrict__ idx_pairs,
    const float* __restrict__ p3b,  const float* __restrict__ f1b,
    const float* __restrict__ f2b,  const float* __restrict__ f3w,
    const float* __restrict__ f3b,
    const _Float16* __restrict__ f1wT, const _Float16* __restrict__ f2wT,
    float* __restrict__ out) {
  __shared__ __align__(16) _Float16 hin[16 * 1024];   // 32768 B
  __shared__ __align__(16) _Float16 hf1[16 * 512];    // 16384 B
  __shared__ __align__(16) float    hf2[16 * 32];     //  2048 B

  const int row0 = blockIdx.x * 16;
  const int tid  = threadIdx.x;
  const int lane = tid & 31;
  const int wave = tid >> 5;

  // segment mean + p3 bias, f32 -> f16
  for (int i = tid; i < 16 * 1024; i += 256) {
    int r = i >> 10, c = i & 1023;
    float cnt = (float)idx_pairs[row0 + r];
    hin[i] = (_Float16)(hacc[(size_t)(row0 + r) * 1024 + c] / cnt + p3b[c]);
  }
  __syncthreads();

  // f1: [16,1024] @ [1024,512] + relu
  for (int nt = wave; nt < 32; nt += 8) {
    v8f acc = {};
    for (int k = 0; k < 1024; k += 32) {
      v16h a = load_a_frag(hin, 1024, 0, k, lane);
      v16h b = load_b_fragT(f1wT, 1024, nt * 16, k, lane);
      acc = wmma_f16(a, b, acc);
    }
    store_relu_f16(hf1, 512, 0, nt * 16, acc, f1b + nt * 16, lane);
  }
  __syncthreads();

  // f2: [16,512] @ [512,32] + relu (f32 out)
  for (int nt = wave; nt < 2; nt += 8) {
    v8f acc = {};
    for (int k = 0; k < 512; k += 32) {
      v16h a = load_a_frag(hf1, 512, 0, k, lane);
      v16h b = load_b_fragT(f2wT, 512, nt * 16, k, lane);
      acc = wmma_f16(a, b, acc);
    }
    int hi = lane >> 4, n = nt * 16 + (lane & 15);
    float bias = f2b[n];
#pragma unroll
    for (int r = 0; r < 8; ++r) {
      float v = acc[r] + bias;
      hf2[(r + 8 * hi) * 32 + n] = v > 0.f ? v : 0.f;
    }
  }
  __syncthreads();

  // f3: [16,32] @ [32,1] — scalar f32 dot per row
  if (tid < 16) {
    float s = f3b[0];
#pragma unroll
    for (int k = 0; k < 32; ++k) s += hf2[tid * 32 + k] * f3w[k];
    out[row0 + tid] = s;
  }
}

// ---------------------------------------------------------------------------
// Launch
// ---------------------------------------------------------------------------
extern "C" void kernel_launch(void* const* d_in, const int* in_sizes, int n_in,
                              void* d_out, int out_size, void* d_ws, size_t ws_size,
                              hipStream_t stream) {
  // Inputs in setup_inputs() order
  const float* pairs = (const float*)d_in[0];
  const int*   idx   = (const int*)  d_in[1];
  // d_in[2] = ref_feats (unused by reference)
  const float* aw  = (const float*)d_in[3];  const float* ab  = (const float*)d_in[4];
  const float* bw  = (const float*)d_in[5];  const float* bb  = (const float*)d_in[6];
  const float* p1w = (const float*)d_in[7];  const float* p1b = (const float*)d_in[8];
  const float* p2w = (const float*)d_in[9];  const float* p2b = (const float*)d_in[10];
  const float* p3w = (const float*)d_in[11]; const float* p3b = (const float*)d_in[12];
  const float* f1w = (const float*)d_in[13]; const float* f1b = (const float*)d_in[14];
  const float* f2w = (const float*)d_in[15]; const float* f2b = (const float*)d_in[16];
  const float* f3w = (const float*)d_in[17]; const float* f3b = (const float*)d_in[18];
  float* out = (float*)d_out;

  const int B = in_sizes[1];                     // 2000 groups

  // Workspace layout: hacc [B][1024] f32, then f16 transposed weights
  float* hacc = (float*)d_ws;
  _Float16* wb = (_Float16*)((char*)d_ws + (size_t)B * 1024 * sizeof(float));
  _Float16* awT  = wb;                  // [128][64]
  _Float16* bwT  = awT  + 128 * 64;     // [128][32]
  _Float16* p1wT = bwT  + 128 * 32;     // [512][384]
  _Float16* p2wT = p1wT + 512 * 384;    // [256][512]
  _Float16* p3wT = p2wT + 256 * 512;    // [1024][256]
  _Float16* f1wT = p3wT + 1024 * 256;   // [512][1024]
  _Float16* f2wT = f1wT + 512 * 1024;   // [32][512]

  // 0) zero accumulator
  {
    int n = B * 1024;
    tgnn_zero_kernel<<<(n + 255) / 256, 256, 0, stream>>>(hacc, n);
  }
  // 1) weight prep (f32 [K,N] -> f16 [N,K])
  auto prep = [&](const float* w, _Float16* wT, int K, int N) {
    int n = K * N;
    tgnn_wprep_kernel<<<(n + 255) / 256, 256, 0, stream>>>(w, wT, K, N);
  };
  prep(aw,  awT,  64,   128);
  prep(bw,  bwT,  32,   128);
  prep(p1w, p1wT, 384,  512);
  prep(p2w, p2wT, 512,  256);
  prep(p3w, p3wT, 256,  1024);
  prep(f1w, f1wT, 1024, 512);
  prep(f2w, f2wT, 512,  32);

  // 2) fused pair MLP + segment sums
  tgnn_pair_mlp_kernel<<<B * CHUNKS, 256, 0, stream>>>(
      pairs, ab, bb, p1b, p2b, awT, bwT, p1wT, p2wT, p3wT, hacc);

  // 3) head MLP -> out [B,1]
  tgnn_head_kernel<<<(B + 15) / 16, 256, 0, stream>>>(
      hacc, idx, p3b, f1b, f2b, f3w, f3b, f1wT, f2wT, out);
}
